// DelayRNN_32418413150940
// MI455X (gfx1250) — matrix operand order repaired and checked by reading, not compiled, not run on hardware
//
#include <hip/hip_runtime.h>

// ---------------------------------------------------------------------------
// DelayRNN on gfx1250 (MI455X): persistent per-batch-slice workgroups,
// bf16 WMMA (v_wmma_f32_16x16x32_bf16), delay buffer resident in LDS.
//
// B=32, S=256, I=128, H=512, C=64, MAX_DELAY=16, BUF_D=17, OUT_LEN=64.
// 4 workgroups x 256 threads (8 waves). Each WG owns 8 batch rows for all
// 320 sequential steps. Weights pre-transposed/converted to bf16 [N][K] in
// workspace so B-fragments are contiguous 128-bit global loads (L2 resident).
// A-staging regions carry one extra all-zero pad row so the M=8->16 padding
// needs no predication: row index is clamped, loads are 2x ds_load_b128.
// K-slice loops are kept rolled (#pragma unroll 1) to avoid VGPR spills;
// latency is hidden across the 8 resident waves.
// ---------------------------------------------------------------------------

typedef __attribute__((ext_vector_type(16))) __bf16 v16bf;
typedef __attribute__((ext_vector_type(8)))  float  v8f;

#define HD    512
#define IN_D  128
#define BLOC  8          // batch rows per workgroup
#define BUFD  17
#define SEQ   256
#define OUTL  64
#define PS    (BLOC*HD)  // buffer plane stride (4096 floats)
#define AROWS (BLOC+1)   // staging rows incl. zero pad row

// ---- A fragment: 16x32 bf16 tile from LDS (row-major, stride ld elems).
// 16-bit A layout (ISA 7.12.2): lane half h, row = lane&15,
// VGPR v holds K pair: K = (v>=4?16:0) + h*8 + (v&3)*2 (+0/+1).
// Per lane this is two contiguous 16-byte runs -> two ds_load_b128.
// Rows >= BLOC are mapped to the all-zero pad row BLOC (no predication).
__device__ __forceinline__ v16bf load_a(const __bf16* p, int ld, int k0, int lane) {
  int lh  = lane >> 4;
  int row = lane & 15;
  row = (row < BLOC) ? row : BLOC;            // pad row is zero
  union { v16bf v; uint4 q[2]; } r;
  const uint4* base = (const uint4*)(p + row * ld + k0 + lh * 8);
  r.q[0] = base[0];                            // K = k0      + lh*8 + 0..7
  r.q[1] = base[2];                            // K = k0 + 16 + lh*8 + 0..7
  return r.v;
}

// ---- B fragment: 32x16 bf16 tile from global transposed weights WT[n][K].
// Lane half h, col=lane&15: K = k0 + h*8 + 0..7 and k0 + 16 + h*8 + 0..7 (matches A).
// Two 8-bf16 runs per lane -> two 128-bit loads.
__device__ __forceinline__ v16bf load_b(const __bf16* wt, int K, int n0, int k0, int lane) {
  int lh = lane >> 4;
  int n  = n0 + (lane & 15);
  const uint4* p = (const uint4*)(wt + (size_t)n * K + k0 + lh * 8);
  union { v16bf v; uint4 q[2]; } r;
  r.q[0] = p[0];
  r.q[1] = p[2];
  return r.v;
}

__device__ __forceinline__ v8f wmma_bf16(v16bf a, v16bf b, v8f c) {
  return __builtin_amdgcn_wmma_f32_16x16x32_bf16(false, a, false, b, (short)0, c,
                                                 false, false);
}

__device__ __forceinline__ float sigmoidf_(float x) {
  return 1.0f / (1.0f + __expf(-x));
}

// ---- weight convert + transpose: src f32 [K][N] -> dst bf16 [N][K]
__global__ void wt_transpose_bf16(const float* __restrict__ src,
                                  __bf16* __restrict__ dst, int K, int N) {
  int idx = blockIdx.x * blockDim.x + threadIdx.x;
  if (idx >= K * N) return;
  int k = idx / N;
  int n = idx - k * N;
  dst[(size_t)n * K + k] = (__bf16)src[idx];
}

// ---------------------------------------------------------------------------
__global__ void __launch_bounds__(256, 1)
delay_rnn_main(const float* __restrict__ x, const int* __restrict__ lengths,
               const __bf16* __restrict__ WinT,  const float* __restrict__ b_in,
               const __bf16* __restrict__ WpassT,const float* __restrict__ b_pass,
               const __bf16* __restrict__ WtauT, const float* __restrict__ b_tau,
               const __bf16* __restrict__ WmemT, const float* __restrict__ b_mem,
               const __bf16* __restrict__ WoutT, const float* __restrict__ b_out,
               float* __restrict__ out) {
  extern __shared__ __align__(16) char smem_raw[];
  float*  buf   = (float*)smem_raw;                 // [17][8][512] f32 delay buffer
  __bf16* hbuf  = (__bf16*)(buf + BUFD * PS);       // [9][512]  h0 (+ zero pad row)
  __bf16* hcur  = hbuf  + AROWS * HD;               // [9][512]  current h (+ pad)
  __bf16* xst   = hcur  + AROWS * HD;               // [9][128]  x_t (+ pad)
  __bf16* taust = xst   + AROWS * IN_D;             // [8][512] tau
  __bf16* memst = taust + BLOC * HD;                // [8][512] mem_l

  const int tid  = threadIdx.x;
  const int lane = tid & 31;
  const int wave = tid >> 5;
  const int lh   = lane >> 4;        // lane half
  const int nloc = lane & 15;
  const int bi   = blockIdx.x;       // batch slice

  // zero LDS state once per launch (buffer + all bf16 staging regions)
  for (int i = tid; i < BUFD * PS; i += 256) buf[i] = 0.0f;
  {
    const int nbf = AROWS * HD * 2 + AROWS * IN_D + BLOC * HD * 2;
    for (int i = tid; i < nbf; i += 256) hbuf[i] = (__bf16)0.0f;
  }

  // mask lengths for my C-tile rows (only lane-half 0 rows are real batches)
  int lenr[8];
  #pragma unroll
  for (int r = 0; r < 8; ++r)
    lenr[r] = (lh == 0) ? lengths[bi * BLOC + r] : 0;

  __syncthreads();

  #pragma unroll 1
  for (int t = 0; t < SEQ + OUTL; ++t) {
    const bool enc = (t < SEQ);
    const int  hp  = t % BUFD;       // physical plane of logical slot 0

    // ---- P0: stage x_t (encode only)
    if (enc) {
      #pragma unroll 1
      for (int i = tid; i < BLOC * IN_D; i += 256) {
        int b = i >> 7, ii = i & (IN_D - 1);
        xst[b * IN_D + ii] =
            (__bf16)x[((size_t)(bi * BLOC + b) * SEQ + t) * IN_D + ii];
      }
    }
    __syncthreads();

    // ---- P1: h = [h0 | x] @ W_in + b_in   (wave owns N-tiles 4w..4w+3)
    v8f ch[4];
    #pragma unroll
    for (int j = 0; j < 4; ++j) {
      float bb = b_in[(wave * 4 + j) * 16 + nloc];
      #pragma unroll
      for (int r = 0; r < 8; ++r) ch[j][r] = bb;
    }
    #pragma unroll 1
    for (int ks = 0; ks < 16; ++ks) {            // h0 part, K = 0..511
      v16bf a = load_a(hbuf, HD, ks * 32, lane);
      #pragma unroll
      for (int j = 0; j < 4; ++j) {
        v16bf b = load_b(WinT, HD + IN_D, (wave * 4 + j) * 16, ks * 32, lane);
        ch[j] = wmma_bf16(a, b, ch[j]);
      }
    }
    if (enc) {                                   // x part, K = 512..639
      #pragma unroll 1
      for (int ks = 0; ks < 4; ++ks) {
        v16bf a = load_a(xst, IN_D, ks * 32, lane);
        #pragma unroll
        for (int j = 0; j < 4; ++j) {
          v16bf b = load_b(WinT, HD + IN_D, (wave * 4 + j) * 16, HD + ks * 32, lane);
          ch[j] = wmma_bf16(a, b, ch[j]);
        }
      }
    } else if (wave < 4) {                       // decode: out = h0 @ W_out + b_out
      v8f co;
      float bb = b_out[wave * 16 + nloc];
      #pragma unroll
      for (int r = 0; r < 8; ++r) co[r] = bb;
      #pragma unroll 1
      for (int ks = 0; ks < 16; ++ks) {
        v16bf a = load_a(hbuf, HD, ks * 32, lane);
        v16bf b = load_b(WoutT, HD, wave * 16, ks * 32, lane);
        co = wmma_bf16(a, b, co);
      }
      if (lh == 0) {
        #pragma unroll
        for (int r = 0; r < 8; ++r)
          out[((size_t)(bi * BLOC + r) * OUTL + (t - SEQ)) * 64 + wave * 16 + nloc] = co[r];
      }
    }

    // stage h -> hcur (bf16); C layout: VGPR r -> row r (lane-half 0), col n
    if (lh == 0) {
      #pragma unroll
      for (int j = 0; j < 4; ++j) {
        int n = (wave * 4 + j) * 16 + nloc;
        #pragma unroll
        for (int r = 0; r < 8; ++r) hcur[r * HD + n] = (__bf16)ch[j][r];
      }
    }
    __syncthreads();

    // ---- P2: p = h @ W_pass + b_pass; blend h = m*p + (1-m)*h (in registers)
    v8f cp[4];
    #pragma unroll
    for (int j = 0; j < 4; ++j) {
      float bb = b_pass[(wave * 4 + j) * 16 + nloc];
      #pragma unroll
      for (int r = 0; r < 8; ++r) cp[j][r] = bb;
    }
    #pragma unroll 1
    for (int ks = 0; ks < 16; ++ks) {
      v16bf a = load_a(hcur, HD, ks * 32, lane);
      #pragma unroll
      for (int j = 0; j < 4; ++j) {
        v16bf b = load_b(WpassT, HD, (wave * 4 + j) * 16, ks * 32, lane);
        cp[j] = wmma_bf16(a, b, cp[j]);
      }
    }
    #pragma unroll
    for (int j = 0; j < 4; ++j)
      #pragma unroll
      for (int r = 0; r < 8; ++r) {
        float m = enc ? ((t < lenr[r]) ? 1.0f : 0.0f) : 1.0f;
        ch[j][r] = m * cp[j][r] + (1.0f - m) * ch[j][r];
      }
    __syncthreads();                  // all GEMM2 reads of hcur done
    if (lh == 0) {
      #pragma unroll
      for (int j = 0; j < 4; ++j) {
        int n = (wave * 4 + j) * 16 + nloc;
        #pragma unroll
        for (int r = 0; r < 8; ++r) hcur[r * HD + n] = (__bf16)ch[j][r];
      }
    }
    __syncthreads();

    // ---- P3: tau / mem_l GEMMs (share A fragment per k-slice)
    v8f ct[4], cm[4];
    #pragma unroll
    for (int j = 0; j < 4; ++j) {
      float bt = b_tau[(wave * 4 + j) * 16 + nloc];
      float bm = b_mem[(wave * 4 + j) * 16 + nloc];
      #pragma unroll
      for (int r = 0; r < 8; ++r) { ct[j][r] = bt; cm[j][r] = bm; }
    }
    #pragma unroll 1
    for (int ks = 0; ks < 16; ++ks) {
      v16bf a = load_a(hcur, HD, ks * 32, lane);
      #pragma unroll
      for (int j = 0; j < 4; ++j) {
        v16bf bt = load_b(WtauT, HD, (wave * 4 + j) * 16, ks * 32, lane);
        ct[j] = wmma_bf16(a, bt, ct[j]);
        v16bf bm = load_b(WmemT, HD, (wave * 4 + j) * 16, ks * 32, lane);
        cm[j] = wmma_bf16(a, bm, cm[j]);
      }
    }
    if (lh == 0) {
      #pragma unroll
      for (int j = 0; j < 4; ++j) {
        int n = (wave * 4 + j) * 16 + nloc;
        #pragma unroll
        for (int r = 0; r < 8; ++r) {
          float tv = 16.0f * sigmoidf_(ct[j][r]);
          tv = fminf(fmaxf(tv, 1.0f), 16.0f);
          taust[r * HD + n] = (__bf16)tv;
          memst[r * HD + n] = (__bf16)sigmoidf_(cm[j][r]);
        }
      }
    }
    __syncthreads();

    // ---- P4: circular delay-buffer update, float4 over hidden dim.
    // logical slot l lives at plane (hp+l)%17; l=1..16 get += w_l*h,
    // old plane hp becomes new logical slot 16 (zeroed). New logical 0
    // (old l=1 result) also feeds hbuf for the next step.
    #pragma unroll 1
    for (int g = tid; g < (BLOC * HD) / 4; g += 256) {
      int b   = g >> 7;
      int j4  = (g & 127) << 2;
      int base = b * HD + j4;
      float tv[4], mv[4], hv[4], h0n[4];
      #pragma unroll
      for (int e = 0; e < 4; ++e) {
        tv[e] = (float)taust[base + e];
        mv[e] = (float)memst[base + e];
        hv[e] = (float)hcur[base + e];
      }
      #pragma unroll
      for (int l = 1; l <= 16; ++l) {
        int ph = hp + l; if (ph >= BUFD) ph -= BUFD;
        float4* pp = (float4*)&buf[ph * PS + base];
        float4 v = *pp;
        #pragma unroll
        for (int e = 0; e < 4; ++e) {
          float w = mv[e] / (1.0f + fabsf(tv[e] - (float)l));
          (&v.x)[e] += w * hv[e];
        }
        *pp = v;
        if (l == 1) {
          #pragma unroll
          for (int e = 0; e < 4; ++e) h0n[e] = (&v.x)[e];
        }
      }
      *(float4*)&buf[hp * PS + base] = make_float4(0.f, 0.f, 0.f, 0.f);
      #pragma unroll
      for (int e = 0; e < 4; ++e) hbuf[base + e] = (__bf16)h0n[e];
    }
    __syncthreads();
  }
}

// ---------------------------------------------------------------------------
extern "C" void kernel_launch(void* const* d_in, const int* in_sizes, int n_in,
                              void* d_out, int out_size, void* d_ws, size_t ws_size,
                              hipStream_t stream) {
  const float* x       = (const float*)d_in[0];
  const int*   lengths = (const int*)d_in[1];
  // d_in[2] = out_lengths (compile-time 64)
  const float* W_in    = (const float*)d_in[3];
  const float* b_in    = (const float*)d_in[4];
  const float* W_pass  = (const float*)d_in[5];
  const float* b_pass  = (const float*)d_in[6];
  const float* W_tau   = (const float*)d_in[7];
  const float* b_tau   = (const float*)d_in[8];
  const float* W_mem   = (const float*)d_in[9];
  const float* b_mem   = (const float*)d_in[10];
  const float* W_out   = (const float*)d_in[11];
  const float* b_out   = (const float*)d_in[12];
  float* out = (float*)d_out;

  // workspace: transposed bf16 weights [N][K]
  __bf16* WinT   = (__bf16*)d_ws;                 // 512 x 640
  __bf16* WpassT = WinT   + (size_t)512 * 640;    // 512 x 512
  __bf16* WtauT  = WpassT + (size_t)512 * 512;
  __bf16* WmemT  = WtauT  + (size_t)512 * 512;
  __bf16* WoutT  = WmemT  + (size_t)512 * 512;    // 64 x 512

  const int THR = 256;
  wt_transpose_bf16<<<(640 * 512 + THR - 1) / THR, THR, 0, stream>>>(W_in,   WinT,   640, 512);
  wt_transpose_bf16<<<(512 * 512 + THR - 1) / THR, THR, 0, stream>>>(W_pass, WpassT, 512, 512);
  wt_transpose_bf16<<<(512 * 512 + THR - 1) / THR, THR, 0, stream>>>(W_tau,  WtauT,  512, 512);
  wt_transpose_bf16<<<(512 * 512 + THR - 1) / THR, THR, 0, stream>>>(W_mem,  WmemT,  512, 512);
  wt_transpose_bf16<<<(512 * 64  + THR - 1) / THR, THR, 0, stream>>>(W_out,  WoutT,  512, 64);

  // dynamic LDS: buffer + (9-row padded) staging regions + tau/mem stages
  size_t smem = (size_t)BUFD * PS * sizeof(float)                     // 278528
              + (size_t)AROWS * HD * sizeof(__bf16) * 2               // hbuf/hcur
              + (size_t)AROWS * IN_D * sizeof(__bf16)                 // xst
              + (size_t)BLOC * HD * sizeof(__bf16) * 2;               // tau/mem
  hipFuncSetAttribute((const void*)delay_rnn_main,
                      hipFuncAttributeMaxDynamicSharedMemorySize, (int)smem);

  delay_rnn_main<<<4, 256, smem, stream>>>(x, lengths,
                                           WinT, b_in, WpassT, b_pass,
                                           WtauT, b_tau, WmemT, b_mem,
                                           WoutT, b_out, out);
}
